// RASH_60395830117193
// MI455X (gfx1250) — compile-verified, matrix operand order checked by reading, not run on hardware
//
#include <hip/hip_runtime.h>
#include <math.h>

typedef __attribute__((ext_vector_type(16))) __bf16 v16bf;
typedef __attribute__((ext_vector_type(2)))  __bf16 v2bf;
typedef __attribute__((ext_vector_type(8)))  float  v8f;

#define N_NODES 4096
#define FDIM    512
#define CDIM    256
#define KPOS    5
#define TAU_INV 1.25f   // 1/0.8

#define BM 128
#define BN 64
#define BK 32

__device__ __forceinline__ __bf16 to_bf16(float f) { return (__bf16)f; }

// permuted LDS fragment index for 16-bit WMMA operands:
//   half(k) = (k>>3)&1 ; pos(k) = (k&7) + (k>=16 ? 8 : 0)
// fragment for (row, half) = 16 contiguous bf16 at base + row*32 + half*16
__device__ __forceinline__ void store_pair_lds(__bf16* rowbase, int k, float lo, float hi)
{
    const int h   = (k >> 3) & 1;
    const int pos = (k & 7) + ((k >> 4) << 3);
    v2bf pr; pr[0] = to_bf16(lo); pr[1] = to_bf16(hi);
    *(v2bf*)(rowbase + h * 16 + pos) = pr;     // 4B-aligned packed store
}

// ---------------------------------------------------------------------------
// GEMM  C[M,N] = A[M,K] @ B[K,N] (+ optional bias[N]); fp32 in/out,
// bf16 WMMA f32-accum, LDS-staged tiles, software-pipelined global loads.
// Block = 256 threads (8 waves); block tile 128x64; wave tile 16x64.
// grid = (N/64, M/128). Requires M%128==0, N%64==0, K%32==0.
// ---------------------------------------------------------------------------
__global__ __launch_bounds__(256)
void gemm_bf16_wmma(const float* __restrict__ A, const float* __restrict__ B,
                    const float* __restrict__ bias, float* __restrict__ C,
                    int M, int N, int K)
{
    __shared__ __align__(32) __bf16 Asm[BM * BK];   // [row][half][pos]
    __shared__ __align__(32) __bf16 Bsm[BN * BK];   // [col][half][pos]

    const int tid  = threadIdx.x;
    const int lane = tid & 31;
    const int wave = tid >> 5;
    const int half = lane >> 4;
    const int lrow = lane & 15;
    const int row0 = blockIdx.y * BM;
    const int col0 = blockIdx.x * BN;

    // A staging map: thread -> (row ar, 16-float chunk asub)
    const int ar   = tid >> 1;
    const int asub = tid & 1;
    const float* Abase = A + (size_t)(row0 + ar) * K + asub * 16;

    // B staging map: two float4s per thread; f = tid*2+j -> (k = f>>4, c4 = f&15)
    const int f0  = tid * 2;
    const int bk0 = f0 >> 4,       bc0 = f0 & 15;
    const int bk1 = (f0 + 1) >> 4, bc1 = (f0 + 1) & 15;
    const float* Bc0 = B + col0 + bc0 * 4;
    const float* Bc1 = B + col0 + bc1 * 4;

    v8f acc0 = {}, acc1 = {}, acc2 = {}, acc3 = {};

    // preload first tile into registers
    float4 a0 = ((const float4*)Abase)[0];
    float4 a1 = ((const float4*)Abase)[1];
    float4 a2 = ((const float4*)Abase)[2];
    float4 a3 = ((const float4*)Abase)[3];
    float4 b0r = *(const float4*)(Bc0 + (size_t)bk0 * N);
    float4 b1r = *(const float4*)(Bc1 + (size_t)bk1 * N);

    for (int k0 = 0; k0 < K; k0 += BK) {
        // ---- stage current tile (registers -> bf16 LDS, permuted) ----
        {
            __bf16* arow = Asm + ar * 32;
            const int kb = asub * 16;
            store_pair_lds(arow, kb + 0,  a0.x, a0.y);
            store_pair_lds(arow, kb + 2,  a0.z, a0.w);
            store_pair_lds(arow, kb + 4,  a1.x, a1.y);
            store_pair_lds(arow, kb + 6,  a1.z, a1.w);
            store_pair_lds(arow, kb + 8,  a2.x, a2.y);
            store_pair_lds(arow, kb + 10, a2.z, a2.w);
            store_pair_lds(arow, kb + 12, a3.x, a3.y);
            store_pair_lds(arow, kb + 14, a3.z, a3.w);

            int h0 = (bk0 >> 3) & 1, p0 = (bk0 & 7) + ((bk0 >> 4) << 3);
            __bf16* bp0 = Bsm + (bc0 * 4) * 32 + h0 * 16 + p0;
            bp0[0] = to_bf16(b0r.x); bp0[32] = to_bf16(b0r.y);
            bp0[64] = to_bf16(b0r.z); bp0[96] = to_bf16(b0r.w);

            int h1 = (bk1 >> 3) & 1, p1 = (bk1 & 7) + ((bk1 >> 4) << 3);
            __bf16* bp1 = Bsm + (bc1 * 4) * 32 + h1 * 16 + p1;
            bp1[0] = to_bf16(b1r.x); bp1[32] = to_bf16(b1r.y);
            bp1[64] = to_bf16(b1r.z); bp1[96] = to_bf16(b1r.w);
        }
        __syncthreads();

        // ---- prefetch next tile into registers (overlaps with WMMA) ----
        if (k0 + BK < K) {
            const float* An = Abase + k0 + BK;
            a0 = ((const float4*)An)[0];
            a1 = ((const float4*)An)[1];
            a2 = ((const float4*)An)[2];
            a3 = ((const float4*)An)[3];
            b0r = *(const float4*)(Bc0 + (size_t)(k0 + BK + bk0) * N);
            b1r = *(const float4*)(Bc1 + (size_t)(k0 + BK + bk1) * N);
        }
        if (k0 + 2 * BK < K) {
            __builtin_prefetch(Abase + k0 + 2 * BK, 0, 1);
            __builtin_prefetch(Bc0 + (size_t)(k0 + 2 * BK + bk0) * N, 0, 1);
        }

        // ---- compute: 1 A-frag + 4 B-frags from LDS, 4 WMMA ----
        {
            const v16bf* Af = (const v16bf*)Asm;
            const v16bf* Bf = (const v16bf*)Bsm;
            v16bf a  = Af[(wave * 16 + lrow) * 2 + half];
            v16bf q0 = Bf[(lrow)      * 2 + half];
            v16bf q1 = Bf[(16 + lrow) * 2 + half];
            v16bf q2 = Bf[(32 + lrow) * 2 + half];
            v16bf q3 = Bf[(48 + lrow) * 2 + half];
            acc0 = __builtin_amdgcn_wmma_f32_16x16x32_bf16(false, a, false, q0, (short)0, acc0, false, false);
            acc1 = __builtin_amdgcn_wmma_f32_16x16x32_bf16(false, a, false, q1, (short)0, acc1, false, false);
            acc2 = __builtin_amdgcn_wmma_f32_16x16x32_bf16(false, a, false, q2, (short)0, acc2, false, false);
            acc3 = __builtin_amdgcn_wmma_f32_16x16x32_bf16(false, a, false, q3, (short)0, acc3, false, false);
        }
        __syncthreads();
    }

    float bb0 = 0.f, bb1 = 0.f, bb2 = 0.f, bb3 = 0.f;
    if (bias) {
        bb0 = bias[col0 + lrow];
        bb1 = bias[col0 + lrow + 16];
        bb2 = bias[col0 + lrow + 32];
        bb3 = bias[col0 + lrow + 48];
    }
    #pragma unroll
    for (int r = 0; r < 8; ++r) {
        size_t row = (size_t)(blockIdx.y * BM + wave * 16 + half * 8 + r);
        float* Cp = C + row * (size_t)N + col0 + lrow;
        Cp[0]  = acc0[r] + bb0;
        Cp[16] = acc1[r] + bb1;
        Cp[32] = acc2[r] + bb2;
        Cp[48] = acc3[r] + bb3;
    }
}

// ---------------------------------------------------------------------------
// Fused contrastive denominator: denom[i] = sum_j exp( (Zn[i]·Hn[j]) / tau )
// WMMA 16x16 score tiles; exp + row-reduction in registers; S never stored.
// ---------------------------------------------------------------------------
__global__ __launch_bounds__(256)
void rowsum_exp_wmma(const float* __restrict__ Zn, const float* __restrict__ Hn,
                     float* __restrict__ denom, int Nn)
{
    const int lane = threadIdx.x & 31;
    const int wave = threadIdx.x >> 5;
    const int half = lane >> 4;
    const int lrow = lane & 15;
    const int row0 = blockIdx.x * 128 + wave * 16;
    const size_t m = (size_t)(row0 + lrow);

    // Preload all 8 K-fragments of this wave's 16 query rows (K=256), vectorized
    v16bf af[8];
    {
        const float* Zp = Zn + m * (size_t)CDIM + half * 8;
        #pragma unroll
        for (int kf = 0; kf < 8; ++kf) {
            const float4* p = (const float4*)(Zp + kf * 32);
            float4 q0 = p[0], q1 = p[1];
            const float4* p2 = (const float4*)(Zp + kf * 32 + 16);
            float4 q2 = p2[0], q3 = p2[1];
            af[kf][0]  = to_bf16(q0.x); af[kf][1]  = to_bf16(q0.y);
            af[kf][2]  = to_bf16(q0.z); af[kf][3]  = to_bf16(q0.w);
            af[kf][4]  = to_bf16(q1.x); af[kf][5]  = to_bf16(q1.y);
            af[kf][6]  = to_bf16(q1.z); af[kf][7]  = to_bf16(q1.w);
            af[kf][8]  = to_bf16(q2.x); af[kf][9]  = to_bf16(q2.y);
            af[kf][10] = to_bf16(q2.z); af[kf][11] = to_bf16(q2.w);
            af[kf][12] = to_bf16(q3.x); af[kf][13] = to_bf16(q3.y);
            af[kf][14] = to_bf16(q3.z); af[kf][15] = to_bf16(q3.w);
        }
    }

    float rs[8] = {0.f, 0.f, 0.f, 0.f, 0.f, 0.f, 0.f, 0.f};
    for (int n0 = 0; n0 < Nn; n0 += 16) {
        v8f acc = {};
        const float* Hrow = Hn + (size_t)(n0 + lrow) * CDIM + half * 8;
        #pragma unroll
        for (int kf = 0; kf < 8; ++kf) {
            const float4* p = (const float4*)(Hrow + kf * 32);
            float4 q0 = p[0], q1 = p[1];
            const float4* p2 = (const float4*)(Hrow + kf * 32 + 16);
            float4 q2 = p2[0], q3 = p2[1];
            v16bf b;
            b[0]  = to_bf16(q0.x); b[1]  = to_bf16(q0.y);
            b[2]  = to_bf16(q0.z); b[3]  = to_bf16(q0.w);
            b[4]  = to_bf16(q1.x); b[5]  = to_bf16(q1.y);
            b[6]  = to_bf16(q1.z); b[7]  = to_bf16(q1.w);
            b[8]  = to_bf16(q2.x); b[9]  = to_bf16(q2.y);
            b[10] = to_bf16(q2.z); b[11] = to_bf16(q2.w);
            b[12] = to_bf16(q3.x); b[13] = to_bf16(q3.y);
            b[14] = to_bf16(q3.z); b[15] = to_bf16(q3.w);
            acc = __builtin_amdgcn_wmma_f32_16x16x32_bf16(false, af[kf], false, b, (short)0, acc, false, false);
        }
        #pragma unroll
        for (int r = 0; r < 8; ++r) rs[r] += __expf(acc[r] * TAU_INV);
    }

    #pragma unroll
    for (int r = 0; r < 8; ++r) {
        float v = rs[r];
        v += __shfl_xor(v, 8, 16);
        v += __shfl_xor(v, 4, 16);
        v += __shfl_xor(v, 2, 16);
        v += __shfl_xor(v, 1, 16);
        if (lrow == 0) denom[row0 + half * 8 + r] = v;
    }
}

// ---------------------------------------------------------------------------
// Small helper kernels
// ---------------------------------------------------------------------------
__global__ void zero_kernel(float* p, int n)
{
    int i = blockIdx.x * blockDim.x + threadIdx.x;
    if (i < n) p[i] = 0.f;
}

__global__ __launch_bounds__(256)
void rowsum_rsqrt(const float* __restrict__ A, float* __restrict__ dinv, int Ncols)
{
    __shared__ float red[256];
    const int i = blockIdx.x;
    float s = 0.f;
    const float4* A4 = (const float4*)(A + (size_t)i * Ncols);
    for (int j = threadIdx.x; j < Ncols / 4; j += 256) {
        float4 q = A4[j];
        s += q.x + q.y + q.z + q.w;
    }
    red[threadIdx.x] = s;
    __syncthreads();
    for (int o = 128; o > 0; o >>= 1) {
        if (threadIdx.x < o) red[threadIdx.x] += red[threadIdx.x + o];
        __syncthreads();
    }
    if (threadIdx.x == 0) {
        float d = red[0];
        dinv[i] = d > 0.f ? rsqrtf(d) : 0.f;
    }
}

__global__ void rowscale_kernel(const float* __restrict__ X, const float* __restrict__ s,
                                float* __restrict__ Y, int total, int F)
{
    int i = blockIdx.x * blockDim.x + threadIdx.x;
    if (i < total) Y[i] = s[i / F] * X[i];
}

__global__ void sub_rowscale_kernel(const float* __restrict__ X, const float* __restrict__ s,
                                    const float* __restrict__ T, float* __restrict__ Y,
                                    int total, int F)
{
    int i = blockIdx.x * blockDim.x + threadIdx.x;
    if (i < total) Y[i] = X[i] - s[i / F] * T[i];
}

__global__ void relu_inplace_kernel(float* __restrict__ X, int n)
{
    int i = blockIdx.x * blockDim.x + threadIdx.x;
    if (i < n) X[i] = fmaxf(X[i], 0.f);
}

__global__ __launch_bounds__(256)
void colstats_kernel(const float* __restrict__ H, float* __restrict__ mv, int Nr, int Ncols)
{
    __shared__ float s1[256], s2[256];
    const int c = blockIdx.x;
    float a = 0.f, b = 0.f;
    for (int r = threadIdx.x; r < Nr; r += 256) {
        float v = H[(size_t)r * Ncols + c];
        a += v; b += v * v;
    }
    s1[threadIdx.x] = a; s2[threadIdx.x] = b;
    __syncthreads();
    for (int o = 128; o > 0; o >>= 1) {
        if (threadIdx.x < o) { s1[threadIdx.x] += s1[threadIdx.x + o]; s2[threadIdx.x] += s2[threadIdx.x + o]; }
        __syncthreads();
    }
    if (threadIdx.x == 0) {
        float mean = s1[0] / (float)Nr;
        mv[c] = mean;
        mv[Ncols + c] = s2[0] / (float)Nr - mean * mean;
    }
}

__global__ void bn_apply_kernel(const float* __restrict__ Hin, const float* __restrict__ mv,
                                const float* __restrict__ g, const float* __restrict__ be,
                                float* __restrict__ Hout, int total, int Ncols, int relu_after)
{
    int i = blockIdx.x * blockDim.x + threadIdx.x;
    if (i >= total) return;
    int c = i % Ncols;
    float y = g[c] * (Hin[i] - mv[c]) * rsqrtf(mv[Ncols + c] + 1e-5f) + be[c];
    if (relu_after) y = fmaxf(y, 0.f);
    Hout[i] = y;
}

__global__ __launch_bounds__(256)
void rownorm_kernel(const float* __restrict__ H, float* __restrict__ Hn)
{
    __shared__ float red[256];
    const size_t base = (size_t)blockIdx.x * CDIM;
    float v = H[base + threadIdx.x];
    red[threadIdx.x] = v * v;
    __syncthreads();
    for (int o = 128; o > 0; o >>= 1) {
        if (threadIdx.x < o) red[threadIdx.x] += red[threadIdx.x + o];
        __syncthreads();
    }
    float norm = sqrtf(red[0]);
    Hn[base + threadIdx.x] = v / (norm + 1e-8f);
}

// top-5 cosine neighbors restricted to adjacency support; one block per row.
__global__ __launch_bounds__(128)
void topk_pos_kernel(const float* __restrict__ Hn, const float* __restrict__ adj,
                     int* __restrict__ posidx, int Nn)
{
    __shared__ __align__(16) float hi[CDIM];
    __shared__ float sv[128 * KPOS];
    __shared__ int   si[128 * KPOS];
    const int i = blockIdx.x;
    for (int k = threadIdx.x; k < CDIM; k += 128) hi[k] = Hn[(size_t)i * CDIM + k];
    __syncthreads();

    float bv[KPOS]; int bi[KPOS];
    #pragma unroll
    for (int t = 0; t < KPOS; ++t) { bv[t] = -3.0e38f; bi[t] = -1; }

    const float4* hi4 = (const float4*)hi;
    for (int j = threadIdx.x; j < Nn; j += 128) {
        if (adj[(size_t)i * Nn + j] > 0.f) {
            const float4* hj4 = (const float4*)(Hn + (size_t)j * CDIM);
            float s = 0.f;
            for (int k = 0; k < CDIM / 4; ++k) {
                float4 a = hi4[k], b = hj4[k];
                s += a.x * b.x + a.y * b.y + a.z * b.z + a.w * b.w;
            }
            if (s > bv[KPOS - 1]) {
                int t = KPOS - 1;
                while (t > 0 && s > bv[t - 1]) { bv[t] = bv[t - 1]; bi[t] = bi[t - 1]; --t; }
                bv[t] = s; bi[t] = j;
            }
        }
    }
    #pragma unroll
    for (int t = 0; t < KPOS; ++t) { sv[threadIdx.x * KPOS + t] = bv[t]; si[threadIdx.x * KPOS + t] = bi[t]; }
    __syncthreads();

    if (threadIdx.x == 0) {
        float gv[KPOS]; int gi[KPOS];
        #pragma unroll
        for (int t = 0; t < KPOS; ++t) { gv[t] = -3.0e38f; gi[t] = -1; }
        for (int e = 0; e < 128 * KPOS; ++e) {
            float s = sv[e];
            if (s > gv[KPOS - 1]) {
                int j = si[e];
                int t = KPOS - 1;
                while (t > 0 && s > gv[t - 1]) { gv[t] = gv[t - 1]; gi[t] = gi[t - 1]; --t; }
                gv[t] = s; gi[t] = j;
            }
        }
        #pragma unroll
        for (int t = 0; t < KPOS; ++t)
            posidx[i * 6 + t] = (gv[t] > -1.0e30f) ? gi[t] : -1;
        posidx[i * 6 + 5] = i;
    }
}

__device__ __forceinline__ float dot256(const float* a, const float* b)
{
    const float4* a4 = (const float4*)a;
    const float4* b4 = (const float4*)b;
    float s = 0.f;
    for (int k = 0; k < CDIM / 4; ++k) {
        float4 u = a4[k], v = b4[k];
        s += u.x * v.x + u.y * v.y + u.z * v.z + u.w * v.w;
    }
    return s;
}

__global__ void loss_kernel(const float* __restrict__ zn, const float* __restrict__ zhn,
                            const float* __restrict__ denom, const int* __restrict__ posidx,
                            float* __restrict__ lossbuf, int Nn)
{
    int i = blockIdx.x * blockDim.x + threadIdx.x;
    if (i >= Nn) return;
    const float* zi = zn + (size_t)i * CDIM;
    float possum = 0.f;
    bool selfIn = false;
    #pragma unroll
    for (int t = 0; t < KPOS; ++t) {
        int j = posidx[i * 6 + t];
        if (j >= 0) {
            if (j == i) selfIn = true;
            possum += __expf(dot256(zi, zhn + (size_t)j * CDIM) * TAU_INV);
        }
    }
    if (!selfIn)
        possum += __expf(dot256(zi, zhn + (size_t)i * CDIM) * TAU_INV);
    lossbuf[i] = -logf(possum / (denom[i] + 1e-8f) + 1e-8f);
}

__global__ __launch_bounds__(256)
void reduce_add_kernel(const float* __restrict__ v, int n, float scale, float* __restrict__ out)
{
    __shared__ float red[256];
    float s = 0.f;
    for (int i = threadIdx.x; i < n; i += 256) s += v[i];
    red[threadIdx.x] = s;
    __syncthreads();
    for (int o = 128; o > 0; o >>= 1) {
        if (threadIdx.x < o) red[threadIdx.x] += red[threadIdx.x + o];
        __syncthreads();
    }
    if (threadIdx.x == 0) out[0] += red[0] * scale;
}

// ---------------------------------------------------------------------------
// Host orchestration
// ---------------------------------------------------------------------------
static inline void launch_gemm(const float* A, const float* B, const float* bias,
                               float* C, int M, int N, int K, hipStream_t stream)
{
    dim3 grid(N / BN, M / BM);
    gemm_bf16_wmma<<<grid, 256, 0, stream>>>(A, B, bias, C, M, N, K);
}

extern "C" void kernel_launch(void* const* d_in, const int* in_sizes, int n_in,
                              void* d_out, int out_size, void* d_ws, size_t ws_size,
                              hipStream_t stream)
{
    const float* x    = (const float*)d_in[0];
    const float* z1   = (const float*)d_in[1];
    const float* adj  = (const float*)d_in[2];
    const float* adj2 = (const float*)d_in[3];
    const float* W_gcn = (const float*)d_in[4];
    const float* W_het = (const float*)d_in[5];
    const float* hpW1 = (const float*)d_in[6];
    const float* hpb1 = (const float*)d_in[7];
    const float* hpg  = (const float*)d_in[8];
    const float* hpbe = (const float*)d_in[9];
    const float* hpW2 = (const float*)d_in[10];
    const float* hpb2 = (const float*)d_in[11];
    const float* tpW1 = (const float*)d_in[12];
    const float* tpb1 = (const float*)d_in[13];
    const float* tpg  = (const float*)d_in[14];
    const float* tpbe = (const float*)d_in[15];
    const float* tpW2 = (const float*)d_in[16];
    const float* tpb2 = (const float*)d_in[17];
    const float* bng  = (const float*)d_in[18];
    const float* bnb  = (const float*)d_in[19];

    const int N = N_NODES, F = FDIM, C = CDIM;

    float* W = (float*)d_ws;
    size_t o = 0;
    float* t1    = W + o; o += (size_t)N * F;
    float* t2    = W + o; o += (size_t)N * F;
    float* t3    = W + o; o += (size_t)N * F;
    float* hbuf  = W + o; o += (size_t)N * C;
    float* h     = W + o; o += (size_t)N * C;
    float* hn    = W + o; o += (size_t)N * C;
    float* p1    = W + o; o += (size_t)N * C;
    float* zh    = W + o; o += (size_t)N * C;
    float* zhn   = W + o; o += (size_t)N * C;
    float* zn    = W + o; o += (size_t)N * C;
    float* mv    = W + o; o += 2 * C;
    float* dinv2 = W + o; o += N;
    float* denom = W + o; o += N;
    float* lossb = W + o; o += N;
    int*   posidx = (int*)(W + o); o += (size_t)N * 6;

    const int EB = 256;
    const int nNF = N * F, nNC = N * C;

    zero_kernel<<<1, 32, 0, stream>>>((float*)d_out, out_size);
    rownorm_kernel<<<N, 256, 0, stream>>>(z1, zn);

    // ======================= homo branch =======================
    launch_gemm(adj, x,  nullptr, t1, N, F, N, stream);
    launch_gemm(adj, t1, nullptr, t2, N, F, N, stream);
    launch_gemm(t2, W_gcn, nullptr, hbuf, N, C, F, stream);
    relu_inplace_kernel<<<(nNC + EB - 1) / EB, EB, 0, stream>>>(hbuf, nNC);
    colstats_kernel<<<C, 256, 0, stream>>>(hbuf, mv, N, C);
    bn_apply_kernel<<<(nNC + EB - 1) / EB, EB, 0, stream>>>(hbuf, mv, bng, bnb, h, nNC, C, 0);
    rownorm_kernel<<<N, 256, 0, stream>>>(h, hn);
    topk_pos_kernel<<<N, 128, 0, stream>>>(hn, adj, posidx, N);
    launch_gemm(h, hpW1, hpb1, p1, N, C, C, stream);
    colstats_kernel<<<C, 256, 0, stream>>>(p1, mv, N, C);
    bn_apply_kernel<<<(nNC + EB - 1) / EB, EB, 0, stream>>>(p1, mv, hpg, hpbe, hbuf, nNC, C, 1);
    launch_gemm(hbuf, hpW2, hpb2, zh, N, C, C, stream);
    rownorm_kernel<<<N, 256, 0, stream>>>(zh, zhn);
    rowsum_exp_wmma<<<N / 128, 256, 0, stream>>>(zn, zhn, denom, N);
    loss_kernel<<<(N + EB - 1) / EB, EB, 0, stream>>>(zn, zhn, denom, posidx, lossb, N);
    reduce_add_kernel<<<1, 256, 0, stream>>>(lossb, N, 1.0f / (float)N, (float*)d_out);

    // ======================= heter branch =======================
    rowsum_rsqrt<<<N, 256, 0, stream>>>(adj2, dinv2, N);
    rowscale_kernel<<<(nNF + EB - 1) / EB, EB, 0, stream>>>(x, dinv2, t1, nNF, F);
    launch_gemm(adj2, t1, nullptr, t2, N, F, N, stream);
    sub_rowscale_kernel<<<(nNF + EB - 1) / EB, EB, 0, stream>>>(x, dinv2, t2, t1, nNF, F);
    rowscale_kernel<<<(nNF + EB - 1) / EB, EB, 0, stream>>>(t1, dinv2, t2, nNF, F);
    launch_gemm(adj2, t2, nullptr, t3, N, F, N, stream);
    sub_rowscale_kernel<<<(nNF + EB - 1) / EB, EB, 0, stream>>>(t1, dinv2, t3, t2, nNF, F);

    launch_gemm(t2, W_het, nullptr, hbuf, N, C, F, stream);
    relu_inplace_kernel<<<(nNC + EB - 1) / EB, EB, 0, stream>>>(hbuf, nNC);
    colstats_kernel<<<C, 256, 0, stream>>>(hbuf, mv, N, C);
    bn_apply_kernel<<<(nNC + EB - 1) / EB, EB, 0, stream>>>(hbuf, mv, bng, bnb, h, nNC, C, 0);
    rownorm_kernel<<<N, 256, 0, stream>>>(h, hn);
    topk_pos_kernel<<<N, 128, 0, stream>>>(hn, adj, posidx, N);
    launch_gemm(h, tpW1, tpb1, p1, N, C, C, stream);
    colstats_kernel<<<C, 256, 0, stream>>>(p1, mv, N, C);
    bn_apply_kernel<<<(nNC + EB - 1) / EB, EB, 0, stream>>>(p1, mv, tpg, tpbe, hbuf, nNC, C, 1);
    launch_gemm(hbuf, tpW2, tpb2, zh, N, C, C, stream);
    rownorm_kernel<<<N, 256, 0, stream>>>(zh, zhn);
    rowsum_exp_wmma<<<N / 128, 256, 0, stream>>>(zn, zhn, denom, N);
    loss_kernel<<<(N + EB - 1) / EB, EB, 0, stream>>>(zn, zhn, denom, posidx, lossb, N);
    reduce_add_kernel<<<1, 256, 0, stream>>>(lossb, N, 1.0f / (float)N, (float*)d_out);

    (void)in_sizes; (void)n_in; (void)ws_size;
}
